// MutiHeadAttention_5806795784398
// MI455X (gfx1250) — matrix operand order repaired and checked by reading, not compiled, not run on hardware
//
#include <hip/hip_runtime.h>
#include <hip/hip_bf16.h>

// ---------------------------------------------------------------------------
// B=8, S=1024, D=512, H=8, dk=64. Softmax over BATCH axis (axis=0).
// Pipeline: f32->bf16 cvt | qkv = x@W_qkv (only cols 0..639 ever used) |
// fused attention with cross-batch softmax | out = attn_out @ W_out.
// Matmuls: v_wmma_f32_16x16x32_bf16. GEMM B operand staged into LDS via
// async global->LDS copies (ASYNCcnt), double-buffered.
// ---------------------------------------------------------------------------

typedef unsigned short u16;
typedef __attribute__((ext_vector_type(16))) __bf16 v16bf;
typedef __attribute__((ext_vector_type(8)))  float  v8f;

struct U32B { uint4 a, b; };
static_assert(sizeof(U32B) == sizeof(v16bf), "frag size mismatch");

__device__ __forceinline__ u16 f32_to_bf16_bits(float f) {
    unsigned u = __builtin_bit_cast(unsigned, f);
    unsigned r = u + 0x7FFFu + ((u >> 16) & 1u);
    return (u16)(r >> 16);
}
__device__ __forceinline__ __bf16 bfbits(u16 x) {
    return __builtin_bit_cast(__bf16, x);
}

__device__ __forceinline__ v8f wmma_bf16(v16bf a, v16bf b, v8f c) {
    return __builtin_amdgcn_wmma_f32_16x16x32_bf16(
        false, a, false, b, (short)0, c, false, false);
}

// ---- async global -> LDS copy (16B per lane), ASYNCcnt-tracked ------------
// The builtin's params are pointers to gcc-vector int4 in AS1 (global) and
// AS3 (LDS); bind the address spaces via typedef qualifiers so the C-style
// cast performs a proper addrspacecast.
typedef int v4i_g __attribute__((__vector_size__(16)));
typedef __attribute__((address_space(1))) v4i_g as1_v4i;
typedef __attribute__((address_space(3))) v4i_g as3_v4i;

#if __has_builtin(__builtin_amdgcn_global_load_async_to_lds_b128)
#define ASYNC_VIA_BUILTIN 1
#else
#define ASYNC_VIA_BUILTIN 0
#endif

__device__ __forceinline__ void async_copy16(const u16* g, u16* l) {
#if ASYNC_VIA_BUILTIN
    __builtin_amdgcn_global_load_async_to_lds_b128(
        (as1_v4i*)g, (as3_v4i*)l, 0, 0);
#else
    unsigned lds_off = (unsigned)(size_t)l;     // LDS aperture: low bits = offset
    asm volatile("global_load_async_to_lds_b128 %0, %1, off"
                 :: "v"(lds_off), "v"(g) : "memory");
#endif
}

__device__ __forceinline__ void wait_async0() {
#if __has_builtin(__builtin_amdgcn_s_wait_asynccnt)
    __builtin_amdgcn_s_wait_asynccnt(0);
#else
    asm volatile("s_wait_asynccnt 0x0" ::: "memory");
#endif
}

// A fragment (16x32 bf16, row-major, leading dim ld).
__device__ __forceinline__ v16bf load_a_frag_g(const u16* tile, int ld, int lane) {
    int row = lane & 15;
    int kb  = (lane >> 4) << 3;
    const u16* q = tile + (size_t)row * ld + kb;
    U32B t;
    t.a = *(const uint4*)(q);
    t.b = *(const uint4*)(q + 16);
    return __builtin_bit_cast(v16bf, t);
}

// B fragment (32x16 bf16, row-major). Lane ln holds column N=ln%16, K in table order.
__device__ __forceinline__ v16bf load_b_frag_g(const u16* tile, int ld, int lane) {
    int n  = lane & 15;
    int kb = (lane >> 4) << 3;
    v16bf r;
    #pragma unroll
    for (int s = 0; s < 16; ++s)
        r[s] = bfbits(tile[(size_t)(kb + (s & 7) + ((s >> 3) << 4)) * ld + n]);
    return r;
}

// ---------------------------------------------------------------------------
__global__ void cvt_f32_bf16(const float* __restrict__ in, u16* __restrict__ out, int n) {
    int i = blockIdx.x * blockDim.x + threadIdx.x;
    if (i < n) out[i] = f32_to_bf16_bits(in[i]);
}

// ---------------------------------------------------------------------------
// bf16 GEMM: C(MxN;ldc) = A(MxK;lda) @ B(KxN;ldb)
// Block: 256 threads = 8 waves as 4(M) x 2(N); wave tile 32x64; block 128x128.
// B k-slab (32x128) staged in LDS via double-buffered async copies.
// ---------------------------------------------------------------------------
template <bool OUT_BF16>
__global__ __launch_bounds__(256) void gemm_bf16_wmma(
    const u16* __restrict__ A, const u16* __restrict__ B, void* __restrict__ Cout,
    int K, int lda, int ldb, int ldc) {
    __shared__ u16 lds_b[2][32][128];        // 2 x 8 KB double buffer

    int lane = threadIdx.x & 31;
    int wave = threadIdx.x >> 5;
    int wm = wave & 3;
    int wn = wave >> 2;
    int m0  = blockIdx.x * 128 + wm * 32;
    int nb0 = blockIdx.y * 128;
    int n0w = wn * 64;                       // wave's N offset inside the slab

    // Each thread copies 2 x 16B chunks of the 32x128 B slab (512 chunks).
    int c0   = threadIdx.x * 2;
    int row0 = c0 >> 4;                      // 16 chunks per row
    int col0 = (c0 & 15) * 8;                // in u16

    // prologue: stage slab for k0 = 0 into buffer 0
    {
        const u16* g = B + (size_t)row0 * ldb + nb0 + col0;
        async_copy16(g,     &lds_b[0][row0][col0]);
        async_copy16(g + 8, &lds_b[0][row0][col0 + 8]);
    }

    v8f acc[2][4] = {};
    for (int k0 = 0; k0 < K; k0 += 32) {
        int cur = (k0 >> 5) & 1;
        wait_async0();
        __syncthreads();                     // slab[cur] visible to all waves

        if (k0 + 32 < K) {                   // stage next slab into other buffer
            const u16* g = B + (size_t)(k0 + 32 + row0) * ldb + nb0 + col0;
            async_copy16(g,     &lds_b[cur ^ 1][row0][col0]);
            async_copy16(g + 8, &lds_b[cur ^ 1][row0][col0 + 8]);
            __builtin_prefetch(A + (size_t)(m0 + lane) * lda + k0 + 32, 0, 1);
        }

        v16bf a0 = load_a_frag_g(A + (size_t)m0 * lda + k0, lda, lane);
        v16bf a1 = load_a_frag_g(A + (size_t)(m0 + 16) * lda + k0, lda, lane);
        v16bf bf0 = load_b_frag_g(&lds_b[cur][0][n0w +  0], 128, lane);
        v16bf bf1 = load_b_frag_g(&lds_b[cur][0][n0w + 16], 128, lane);
        v16bf bf2 = load_b_frag_g(&lds_b[cur][0][n0w + 32], 128, lane);
        v16bf bf3 = load_b_frag_g(&lds_b[cur][0][n0w + 48], 128, lane);
        acc[0][0] = wmma_bf16(a0, bf0, acc[0][0]);
        acc[0][1] = wmma_bf16(a0, bf1, acc[0][1]);
        acc[0][2] = wmma_bf16(a0, bf2, acc[0][2]);
        acc[0][3] = wmma_bf16(a0, bf3, acc[0][3]);
        acc[1][0] = wmma_bf16(a1, bf0, acc[1][0]);
        acc[1][1] = wmma_bf16(a1, bf1, acc[1][1]);
        acc[1][2] = wmma_bf16(a1, bf2, acc[1][2]);
        acc[1][3] = wmma_bf16(a1, bf3, acc[1][3]);
        // next iteration's top barrier protects slab[cur] from the copy that
        // will be issued into it one iteration later
    }

    #pragma unroll
    for (int i = 0; i < 2; ++i) {
        #pragma unroll
        for (int j = 0; j < 4; ++j) {
            int mb = m0 + i * 16 + ((lane >> 4) << 3);
            int nb = nb0 + n0w + j * 16 + (lane & 15);
            #pragma unroll
            for (int r = 0; r < 8; ++r) {
                float v = acc[i][j][r];
                if (OUT_BF16)
                    ((u16*)Cout)[(size_t)(mb + r) * ldc + nb] = f32_to_bf16_bits(v);
                else
                    ((float*)Cout)[(size_t)(mb + r) * ldc + nb] = v;
            }
        }
    }
}

// ---------------------------------------------------------------------------
// Fused attention with batch-axis softmax.
// Grid: x = S/32 q-tiles, y = H. Block: 256 threads, wave w = batch w.
// qkv: [B][S][640] bf16. Q cols=h*64, K cols=64+h*64, V cols=128+h*64.
// ---------------------------------------------------------------------------
__global__ __launch_bounds__(256) void attention_bsoftmax(
    const u16* __restrict__ qkv, u16* __restrict__ attn_out) {
    const int LD = 640;
    const int S  = 1024;
    int lane = threadIdx.x & 31;
    int b    = threadIdx.x >> 5;
    int h    = blockIdx.y;
    int q0   = blockIdx.x * 32;

    const size_t bstride = (size_t)S * LD;
    const u16* Qp = qkv + (size_t)b * bstride + h * 64;
    const u16* Kp = qkv + (size_t)b * bstride + 64 + h * 64;
    const u16* Vp = qkv + (size_t)b * bstride + 128 + h * 64;

    __shared__ float lds_s[32][32][8];   // scores^T, batch contiguous  32 KB
    __shared__ u16   lds_p[8][32][32];   // attn [b][q][kk] bf16        16 KB

    // 4 loop-invariant Q^T B-fragments (1/sqrt(dk) folded in; exact in bf16)
    v16bf qt[2][2];
    #pragma unroll
    for (int dd = 0; dd < 2; ++dd) {
        #pragma unroll
        for (int qs = 0; qs < 2; ++qs) {
            #pragma unroll
            for (int t = 0; t < 16; ++t) {
                u16 raw = Qp[(size_t)(q0 + qs * 16 + (lane & 15)) * LD + dd * 32 + ((lane >> 4) << 3) + (t & 7) + ((t >> 3) << 4)];
                float f = (float)bfbits(raw) * 0.125f;
                qt[dd][qs][t] = bfbits(f32_to_bf16_bits(f));
            }
        }
    }

    v8f oacc[2][4] = {};

    for (int kk0 = 0; kk0 < S; kk0 += 32) {
        if (kk0 + 32 < S) {
            __builtin_prefetch(Kp + (size_t)(kk0 + 32 + lane) * LD, 0, 1);
            __builtin_prefetch(Vp + (size_t)(kk0 + 32 + lane) * LD, 0, 1);
        }

        // ---- all 4 K A-fragments up-front (one clause, max overlap) ----
        v16bf ka[2][2];   // [kkSub][dd]
        #pragma unroll
        for (int ks = 0; ks < 2; ++ks)
            #pragma unroll
            for (int dd = 0; dd < 2; ++dd)
                ka[ks][dd] = load_a_frag_g(
                    Kp + (size_t)(kk0 + ks * 16) * LD + dd * 32, LD, lane);

        // ---- scores^T (kk32 x q32) = K_tile @ (scaled Q)^T ----
        v8f sT[2][2] = {};
        #pragma unroll
        for (int dd = 0; dd < 2; ++dd) {
            sT[0][0] = wmma_bf16(ka[0][dd], qt[dd][0], sT[0][0]);
            sT[0][1] = wmma_bf16(ka[0][dd], qt[dd][1], sT[0][1]);
            sT[1][0] = wmma_bf16(ka[1][dd], qt[dd][0], sT[1][0]);
            sT[1][1] = wmma_bf16(ka[1][dd], qt[dd][1], sT[1][1]);
        }

        // ---- publish: lds_s[kk][q][b] (batch contiguous) ----
        #pragma unroll
        for (int i = 0; i < 2; ++i) {
            #pragma unroll
            for (int j = 0; j < 2; ++j) {
                int kkb = i * 16 + ((lane >> 4) << 3);
                int qn  = j * 16 + (lane & 15);
                #pragma unroll
                for (int r = 0; r < 8; ++r)
                    lds_s[kkb + r][qn][b] = sT[i][j][r];
            }
        }

        // ---- issue V B-fragment loads early; consumed after softmax ----
        v16bf vb[4];
        #pragma unroll
        for (int j = 0; j < 4; ++j)
            vb[j] = load_b_frag_g(Vp + (size_t)kk0 * LD + j * 16, LD, lane);

        __syncthreads();

        // ---- batch softmax: 1024 elements over 256 threads ----
        #pragma unroll
        for (int e = threadIdx.x; e < 1024; e += 256) {
            int kkr = e >> 5;
            int qn  = e & 31;
            const float* sp = &lds_s[kkr][qn][0];
            float4 lo = *(const float4*)(sp);
            float4 hi = *(const float4*)(sp + 4);
            float s[8] = { lo.x, lo.y, lo.z, lo.w, hi.x, hi.y, hi.z, hi.w };
            float mx = s[0];
            #pragma unroll
            for (int bb = 1; bb < 8; ++bb) mx = fmaxf(mx, s[bb]);
            float ex[8];
            float sum = 0.f;
            #pragma unroll
            for (int bb = 0; bb < 8; ++bb) { ex[bb] = __expf(s[bb] - mx); sum += ex[bb]; }
            float inv = 1.0f / sum;
            #pragma unroll
            for (int bb = 0; bb < 8; ++bb)
                lds_p[bb][qn][kkr] = f32_to_bf16_bits(ex[bb] * inv);
        }
        __syncthreads();

        // ---- out(32q x 64d) += attn(32x32) @ V(32x64) ----
        v16bf pa0, pa1;
        {
            int row = lane & 15;
            int kb  = (lane >> 4) << 3;
            const u16* p0 = &lds_p[b][row][0];
            const u16* p1 = &lds_p[b][16 + row][0];
            U32B t0, t1;
            t0.a = *(const uint4*)(p0 + kb); t0.b = *(const uint4*)(p0 + kb + 16);
            t1.a = *(const uint4*)(p1 + kb); t1.b = *(const uint4*)(p1 + kb + 16);
            pa0 = __builtin_bit_cast(v16bf, t0);
            pa1 = __builtin_bit_cast(v16bf, t1);
        }
        #pragma unroll
        for (int j = 0; j < 4; ++j) {
            oacc[0][j] = wmma_bf16(pa0, vb[j], oacc[0][j]);
            oacc[1][j] = wmma_bf16(pa1, vb[j], oacc[1][j]);
        }
    }

    u16* op = attn_out + (size_t)b * S * 512 + (size_t)q0 * 512 + h * 64;
    #pragma unroll
    for (int i = 0; i < 2; ++i) {
        #pragma unroll
        for (int j = 0; j < 4; ++j) {
            int qb = i * 16 + ((lane >> 4) << 3);
            int dn = j * 16 + (lane & 15);
            #pragma unroll
            for (int r = 0; r < 8; ++r)
                op[(size_t)(qb + r) * 512 + dn] = f32_to_bf16_bits(oacc[i][j][r]);
        }
    }
}

// ---------------------------------------------------------------------------
extern "C" void kernel_launch(void* const* d_in, const int* in_sizes, int n_in,
                              void* d_out, int out_size, void* d_ws, size_t ws_size,
                              hipStream_t stream) {
    (void)in_sizes; (void)n_in; (void)out_size; (void)ws_size;
    const int B = 8, S = 1024, D = 512;
    const int M = B * S;
    const int LDW = 3 * D;               // 1536
    const int NQKV = 640;

    const float* x    = (const float*)d_in[0];
    const float* Wqkv = (const float*)d_in[1];
    const float* Wout = (const float*)d_in[2];
    float*       out  = (float*)d_out;

    u16* xb    = (u16*)d_ws;
    u16* wqkvb = xb    + (size_t)M * D;
    u16* woutb = wqkvb + (size_t)D * LDW;
    u16* qkvb  = woutb + (size_t)D * D;
    u16* aob   = qkvb  + (size_t)M * NQKV;

    {
        int n1 = M * D, n2 = D * LDW, n3 = D * D;
        cvt_f32_bf16<<<(n1 + 255) / 256, 256, 0, stream>>>(x, xb, n1);
        cvt_f32_bf16<<<(n2 + 255) / 256, 256, 0, stream>>>(Wqkv, wqkvb, n2);
        cvt_f32_bf16<<<(n3 + 255) / 256, 256, 0, stream>>>(Wout, woutb, n3);
    }

    {   // qkv[:, 0:640] = x @ W_qkv[:, 0:640]
        dim3 grid(M / 128, NQKV / 128);
        gemm_bf16_wmma<true><<<grid, 256, 0, stream>>>(
            xb, wqkvb, (void*)qkvb, D, D, LDW, NQKV);
    }
    {   // fused attention
        dim3 grid(S / 32, 8);
        attention_bsoftmax<<<grid, 256, 0, stream>>>(qkvb, aob);
    }
    {   // out = attn_out @ W_out
        dim3 grid(M / 128, D / 128);
        gemm_bf16_wmma<false><<<grid, 256, 0, stream>>>(
            aob, woutb, (void*)out, D, D, D, D);
    }
}
